// Attention_6158983102991
// MI455X (gfx1250) — matrix operand order repaired and checked by reading, not compiled, not run on hardware
//
#include <hip/hip_runtime.h>
#include <hip/hip_bf16.h>

// ---------------------------------------------------------------------------
// fastmax (p=1) causal linear attention, chunked-scan formulation, CDNA5 WMMA
//   B=1, N=2048, D_MODEL=1024, H=16, D_HEAD=64, A0=A1=LIM=1
// ---------------------------------------------------------------------------

typedef __bf16 bf16_t;
typedef bf16_t bf16x16 __attribute__((ext_vector_type(16)));
typedef float  f32x8   __attribute__((ext_vector_type(8)));

union FragU { uint4 u4[2]; bf16x16 b; };

__device__ __forceinline__ unsigned short f2bf(float f) {
    union { float f; unsigned u; } v; v.f = f;
    unsigned r = v.u + 0x7FFFu + ((v.u >> 16) & 1u);   // round-to-nearest-even
    return (unsigned short)(r >> 16);
}
__device__ __forceinline__ float bf2f(unsigned short h) {
    union { float f; unsigned u; } v; v.u = ((unsigned)h) << 16;
    return v.f;
}

// ---- CDNA5 async global -> LDS copy (ASYNCcnt-tracked, 16B per lane) -------
__device__ __forceinline__ void async_copy_b128(void* lds_ptr, const void* gptr) {
    // low 32 bits of a generic pointer into LDS == wave-relative LDS offset
    unsigned loff = (unsigned)(size_t)lds_ptr;
    asm volatile("global_load_async_to_lds_b128 %0, %1, off"
                 :: "v"(loff), "v"(gptr) : "memory");
}
__device__ __forceinline__ void wait_async_le2() {
    asm volatile("s_wait_asynccnt 0x2" ::: "memory");
}
__device__ __forceinline__ void wait_async_0() {
    asm volatile("s_wait_asynccnt 0x0" ::: "memory");
}

// A fragment: 16x32 bf16, rows = row_base + lane&15.
// lanes 0-15 hold K {k..k+7, k+16..k+23}; lanes 16-31 hold {k+8..k+15, k+24..k+31}
__device__ __forceinline__ bf16x16 load_fragA(const unsigned short* lds,
                                              int row_base, int k_base, int stride) {
    int lane = threadIdx.x & 31;
    int r = row_base + (lane & 15);
    int k = k_base + ((lane & 16) ? 8 : 0);
    FragU f;
    f.u4[0] = *(const uint4*)(lds + r * stride + k);
    f.u4[1] = *(const uint4*)(lds + r * stride + k + 16);
    return f.b;
}

// B fragment: 32x16 bf16, tile stored as T[n][k] (column n of B = row n of T).
// lanes 0-15 hold K {k..k+7, k+16..k+23}; lanes 16-31 hold {k+8..k+15, k+24..k+31}
__device__ __forceinline__ bf16x16 load_fragB(const unsigned short* lds,
                                              int col_base, int k_base, int stride) {
    int lane = threadIdx.x & 31;
    int n = col_base + (lane & 15);
    int k = k_base + ((lane & 16) ? 8 : 0);
    FragU f;
    f.u4[0] = *(const uint4*)(lds + n * stride + k);
    f.u4[1] = *(const uint4*)(lds + n * stride + k + 16);
    return f.b;
}

__device__ __forceinline__ f32x8 wmma_bf16(bf16x16 a, bf16x16 b, f32x8 c) {
    return __builtin_amdgcn_wmma_f32_16x16x32_bf16(false, a, false, b,
                                                   (short)0, c, false, false);
}

// ---------------------------------------------------------------------------
// Kernel 0: single fused fp32 -> bf16 convert of X, Wq, Wk, Wv, Wo (6M elems)
// ---------------------------------------------------------------------------
__global__ __launch_bounds__(256) void cvt_all(const float* __restrict__ X,
                                               const float* __restrict__ Wq,
                                               const float* __restrict__ Wk,
                                               const float* __restrict__ Wv,
                                               const float* __restrict__ Wo,
                                               unsigned short* __restrict__ Xb,
                                               unsigned short* __restrict__ Wqb,
                                               unsigned short* __restrict__ Wkb,
                                               unsigned short* __restrict__ Wvb,
                                               unsigned short* __restrict__ Wob) {
    const size_t NX = (size_t)2048 * 1024, NW = (size_t)1024 * 1024;
    size_t i = (size_t)blockIdx.x * 256 + threadIdx.x;
    const float* s; unsigned short* d; size_t off;
    if (i < NX)               { s = X;  d = Xb;  off = i; }
    else if (i < NX + NW)     { s = Wq; d = Wqb; off = i - NX; }
    else if (i < NX + 2 * NW) { s = Wk; d = Wkb; off = i - NX - NW; }
    else if (i < NX + 3 * NW) { s = Wv; d = Wvb; off = i - NX - 2 * NW; }
    else                      { s = Wo; d = Wob; off = i - NX - 3 * NW; }
    d[off] = f2bf(s[off]);
}

// ---------------------------------------------------------------------------
// Kernel 1: QKV projection. C = X @ W^T, 64x64 block tile, async double-buffer
// K staging. Output scattered head-major: D[h][token][d] fp32.
// grid (32, 16, 3), 256 threads
// ---------------------------------------------------------------------------
__global__ __launch_bounds__(256) void gemm_qkv(const unsigned short* __restrict__ Xb,
                                                const unsigned short* __restrict__ Wqb,
                                                const unsigned short* __restrict__ Wkb,
                                                const unsigned short* __restrict__ Wvb,
                                                float* __restrict__ Qf,
                                                float* __restrict__ Kf,
                                                float* __restrict__ Vf) {
    int mb = blockIdx.x, h = blockIdx.y, z = blockIdx.z;
    const unsigned short* W = (z == 0) ? Wqb : (z == 1) ? Wkb : Wvb;
    float* D = (z == 0) ? Qf : (z == 1) ? Kf : Vf;

    __shared__ unsigned short Xs[2][64 * 32];
    __shared__ unsigned short Ws[2][64 * 32];

    int tid = threadIdx.x, lane = tid & 31, w = tid >> 5;
    int ni = w & 3, mi0 = w >> 2;            // wave does tiles (mi0,ni) and (mi0+2,ni)
    f32x8 acc[2] = {};

    int lrow = tid >> 2, lcol = (tid & 3) * 8;   // 16B slice per thread
    const unsigned short* xsrc = Xb + (size_t)(mb * 64 + lrow) * 1024 + lcol;
    const unsigned short* wsrc = W  + (size_t)(h  * 64 + lrow) * 1024 + lcol;

    // prologue: stage kb=0 into buffer 0
    async_copy_b128(&Xs[0][lrow * 32 + lcol], xsrc);
    async_copy_b128(&Ws[0][lrow * 32 + lcol], wsrc);

    for (int kb = 0; kb < 1024; kb += 32) {
        int cur = (kb >> 5) & 1;
        if (kb + 32 < 1024) {                 // stage next slab into other buffer
            async_copy_b128(&Xs[cur ^ 1][lrow * 32 + lcol], xsrc + kb + 32);
            async_copy_b128(&Ws[cur ^ 1][lrow * 32 + lcol], wsrc + kb + 32);
            wait_async_le2();                 // current slab's 2 ops retired
        } else {
            wait_async_0();
        }
        __syncthreads();
        bf16x16 b  = load_fragB(Ws[cur], ni * 16, 0, 32);
        bf16x16 a0 = load_fragA(Xs[cur], mi0 * 16, 0, 32);
        acc[0] = wmma_bf16(a0, b, acc[0]);
        bf16x16 a1 = load_fragA(Xs[cur], (mi0 + 2) * 16, 0, 32);
        acc[1] = wmma_bf16(a1, b, acc[1]);
        __syncthreads();                      // all reads of `cur` done before reuse
    }
    for (int ts = 0; ts < 2; ++ts) {
        int mi = mi0 + ts * 2;
        for (int r = 0; r < 8; ++r) {
            int token = mb * 64 + mi * 16 + r + ((lane & 16) ? 8 : 0);
            int d = ni * 16 + (lane & 15);
            D[((size_t)h * 2048 + token) * 64 + d] = acc[ts][r];
        }
    }
}

// ---------------------------------------------------------------------------
// Kernel 2: per-head fastmax scales: s = LIM / max_i ||row_i||   (16 blocks)
// scales[0..15] = q scale, scales[16..31] = k scale
// ---------------------------------------------------------------------------
__global__ __launch_bounds__(256) void norm_scales(const float* __restrict__ Qf,
                                                   const float* __restrict__ Kf,
                                                   float* __restrict__ scales) {
    int h = blockIdx.x, tid = threadIdx.x;
    __shared__ float red[256];
    for (int which = 0; which < 2; ++which) {
        const float* p = (which ? Kf : Qf) + (size_t)h * 2048 * 64;
        float mx = 0.f;
        for (int r = tid; r < 2048; r += 256) {
            const float4* row = (const float4*)(p + (size_t)r * 64);
            float n2 = 0.f;
            for (int d4 = 0; d4 < 16; ++d4) {
                float4 x = row[d4];
                n2 += x.x * x.x + x.y * x.y + x.z * x.z + x.w * x.w;
            }
            mx = fmaxf(mx, n2);
        }
        red[tid] = mx;
        __syncthreads();
        for (int s = 128; s > 0; s >>= 1) {
            if (tid < s) red[tid] = fmaxf(red[tid], red[tid + s]);
            __syncthreads();
        }
        if (tid == 0) scales[which * 16 + h] = rsqrtf(red[0]);   // LIM = 1
        __syncthreads();
    }
}

// ---------------------------------------------------------------------------
// Kernel 3: per-(head,chunk) partials: S_c = K_c^T V_c (64x64), ksum_c, vsum_c
// k pre-scaled by per-head scale.  grid (32 chunks, 16 heads), 256 threads
// ---------------------------------------------------------------------------
__global__ __launch_bounds__(256) void chunk_sums(const float* __restrict__ Kf,
                                                  const float* __restrict__ Vf,
                                                  const float* __restrict__ scales,
                                                  float* __restrict__ Ssum,
                                                  float* __restrict__ ksum,
                                                  float* __restrict__ vsum) {
    int c = blockIdx.x, h = blockIdx.y, tid = threadIdx.x;
    __shared__ float kc[64 * 64];
    __shared__ float vc[64 * 64];
    float ks = scales[16 + h];
    const float* kp = Kf + ((size_t)h * 2048 + c * 64) * 64;
    const float* vp = Vf + ((size_t)h * 2048 + c * 64) * 64;
    for (int i = tid; i < 4096; i += 256) { kc[i] = kp[i] * ks; vc[i] = vp[i]; }
    __syncthreads();

    int d = tid >> 2, e0 = (tid & 3) * 16;
    float acc[16];
#pragma unroll
    for (int i = 0; i < 16; ++i) acc[i] = 0.f;
    for (int t = 0; t < 64; ++t) {
        float kd = kc[t * 64 + d];
#pragma unroll
        for (int i = 0; i < 16; ++i) acc[i] += kd * vc[t * 64 + e0 + i];
    }
    float* Sp = Ssum + ((size_t)h * 32 + c) * 4096;
#pragma unroll
    for (int i = 0; i < 16; ++i) Sp[d * 64 + e0 + i] = acc[i];

    if (tid < 64) {
        float s = 0.f;
        for (int t = 0; t < 64; ++t) s += kc[t * 64 + tid];
        ksum[((size_t)h * 32 + c) * 64 + tid] = s;
    } else if (tid < 128) {
        int e = tid - 64;
        float s = 0.f;
        for (int t = 0; t < 64; ++t) s += vc[t * 64 + e];
        vsum[((size_t)h * 32 + c) * 64 + e] = s;
    }
}

// ---------------------------------------------------------------------------
// Kernel 4: in-place exclusive prefix over the 32 chunks, per head (16 blocks)
// ---------------------------------------------------------------------------
__global__ __launch_bounds__(256) void chunk_prefix(float* __restrict__ Ssum,
                                                    float* __restrict__ ksum,
                                                    float* __restrict__ vsum) {
    int h = blockIdx.x, tid = threadIdx.x;
    for (int idx = tid; idx < 4096; idx += 256) {
        float acc = 0.f;
        float* b = Ssum + (size_t)h * 32 * 4096 + idx;
        for (int c = 0; c < 32; ++c) { float t = b[(size_t)c * 4096]; b[(size_t)c * 4096] = acc; acc += t; }
    }
    if (tid < 64) {
        float acc = 0.f;
        float* b = ksum + (size_t)h * 32 * 64 + tid;
        for (int c = 0; c < 32; ++c) { float t = b[c * 64]; b[c * 64] = acc; acc += t; }
        acc = 0.f;
        b = vsum + (size_t)h * 32 * 64 + tid;
        for (int c = 0; c < 32; ++c) { float t = b[c * 64]; b[c * 64] = acc; acc += t; }
    }
}

// ---------------------------------------------------------------------------
// Kernel 5: per-(head,chunk) output:
//   num = Q_c @ S_prev + Pm @ V_c + vsum_prev,  Pm[i][j] = (j<=i) ? 1 + q_i.k_j : 0
//   den[i] = 64*c + q_i . ksum_prev + rowsum(Pm[i])
//   out = num / den  -> bf16 [token][h*64+e]     grid (32, 16), 256 threads
// ---------------------------------------------------------------------------
__global__ __launch_bounds__(256) void attn_chunk(const float* __restrict__ Qf,
                                                  const float* __restrict__ Kf,
                                                  const float* __restrict__ Vf,
                                                  const float* __restrict__ Sp,
                                                  const float* __restrict__ ksp,
                                                  const float* __restrict__ vsp,
                                                  const float* __restrict__ scales,
                                                  unsigned short* __restrict__ attn_out) {
    int c = blockIdx.x, h = blockIdx.y, tid = threadIdx.x, lane = tid & 31;
    __shared__ unsigned short qb[64 * 64];   // scaled q, bf16, [i][d]
    __shared__ unsigned short kb[64 * 64];   // scaled k, bf16, [j][d]
    __shared__ unsigned short vt[64 * 64];   // v transposed, [e][j]
    __shared__ unsigned short st[64 * 64];   // S_prev transposed, [e][d]
    __shared__ unsigned short pm[64 * 64];   // masked scores, [i][j]
    __shared__ float ksum_p[64], vsum_p[64], den[64];

    float qs = scales[h], ks = scales[16 + h];
    const float* qp = Qf + ((size_t)h * 2048 + c * 64) * 64;
    const float* kp = Kf + ((size_t)h * 2048 + c * 64) * 64;
    const float* vp = Vf + ((size_t)h * 2048 + c * 64) * 64;
    const float* sp = Sp + ((size_t)h * 32 + c) * 4096;

    for (int i = tid; i < 4096; i += 256) {
        int t = i >> 6, d = i & 63;
        qb[i] = f2bf(qp[i] * qs);
        kb[i] = f2bf(kp[i] * ks);
        vt[d * 64 + t] = f2bf(vp[i]);       // vt[e][token]
        st[d * 64 + t] = f2bf(sp[i]);       // sp is [d][e] -> st[e][d]
    }
    if (tid < 64) {
        ksum_p[tid] = ksp[((size_t)h * 32 + c) * 64 + tid];
        vsum_p[tid] = vsp[((size_t)h * 32 + c) * 64 + tid];
    }
    __syncthreads();

    int w = tid >> 5;
    int ni = w & 3, mi0 = w >> 2;            // tiles (mi0,ni),(mi0+2,ni)

    // acc = Q @ S_prev   (Kdim = d, B columns e = st rows)
    f32x8 acc[2] = {};
    for (int kk = 0; kk < 64; kk += 32) {
        bf16x16 b = load_fragB(st, ni * 16, kk, 64);
        bf16x16 a = load_fragA(qb, mi0 * 16, kk, 64);
        acc[0] = wmma_bf16(a, b, acc[0]);
        a = load_fragA(qb, (mi0 + 2) * 16, kk, 64);
        acc[1] = wmma_bf16(a, b, acc[1]);
    }
    // P = Q @ K^T   (Kdim = d, B columns j = kb rows)
    f32x8 p[2] = {};
    for (int kk = 0; kk < 64; kk += 32) {
        bf16x16 b = load_fragB(kb, ni * 16, kk, 64);
        bf16x16 a = load_fragA(qb, mi0 * 16, kk, 64);
        p[0] = wmma_bf16(a, b, p[0]);
        a = load_fragA(qb, (mi0 + 2) * 16, kk, 64);
        p[1] = wmma_bf16(a, b, p[1]);
    }
    // causal mask + a0, write Pm to LDS (C layout -> [i][j])
    int colj = ni * 16 + (lane & 15);
    for (int ts = 0; ts < 2; ++ts) {
        int mi = mi0 + ts * 2;
#pragma unroll
        for (int r = 0; r < 8; ++r) {
            int row = mi * 16 + r + ((lane & 16) ? 8 : 0);
            float val = (colj <= row) ? (1.0f + p[ts][r]) : 0.0f;   // A0 + A1*qk
            pm[row * 64 + colj] = f2bf(val);
        }
    }
    __syncthreads();

    // den per row
    if (tid < 64) {
        float dv = (float)(c * 64);                         // a0 * (#prev tokens)
        for (int dd = 0; dd < 64; ++dd) dv += bf2f(qb[tid * 64 + dd]) * ksum_p[dd];
        float rs = 0.f;
        for (int j = 0; j < 64; ++j) rs += bf2f(pm[tid * 64 + j]);
        den[tid] = dv + rs;
    }
    __syncthreads();

    // acc += Pm @ V   (Kdim = j, B columns e = vt rows)
    for (int kk = 0; kk < 64; kk += 32) {
        bf16x16 b = load_fragB(vt, ni * 16, kk, 64);
        bf16x16 a = load_fragA(pm, mi0 * 16, kk, 64);
        acc[0] = wmma_bf16(a, b, acc[0]);
        a = load_fragA(pm, (mi0 + 2) * 16, kk, 64);
        acc[1] = wmma_bf16(a, b, acc[1]);
    }

    for (int ts = 0; ts < 2; ++ts) {
        int mi = mi0 + ts * 2;
#pragma unroll
        for (int r = 0; r < 8; ++r) {
            int row = mi * 16 + r + ((lane & 16) ? 8 : 0);
            int e   = ni * 16 + (lane & 15);
            float num = acc[ts][r] + vsum_p[e];
            float o = num / den[row];
            attn_out[(size_t)(c * 64 + row) * 1024 + h * 64 + e] = f2bf(o);
        }
    }
}

// ---------------------------------------------------------------------------
// Kernel 6: output projection: out = attn @ Wo^T + bo, async double-buffered
// grid (32,16), 256 threads
// ---------------------------------------------------------------------------
__global__ __launch_bounds__(256) void gemm_out(const unsigned short* __restrict__ Ab,
                                                const unsigned short* __restrict__ Wob,
                                                const float* __restrict__ bo,
                                                float* __restrict__ Out) {
    int mb = blockIdx.x, nb = blockIdx.y;
    __shared__ unsigned short As[2][64 * 32];
    __shared__ unsigned short Ws[2][64 * 32];

    int tid = threadIdx.x, lane = tid & 31, w = tid >> 5;
    int ni = w & 3, mi0 = w >> 2;
    f32x8 acc[2] = {};

    int lrow = tid >> 2, lcol = (tid & 3) * 8;
    const unsigned short* asrc = Ab  + (size_t)(mb * 64 + lrow) * 1024 + lcol;
    const unsigned short* wsrc = Wob + (size_t)(nb * 64 + lrow) * 1024 + lcol;

    async_copy_b128(&As[0][lrow * 32 + lcol], asrc);
    async_copy_b128(&Ws[0][lrow * 32 + lcol], wsrc);

    for (int kb = 0; kb < 1024; kb += 32) {
        int cur = (kb >> 5) & 1;
        if (kb + 32 < 1024) {
            async_copy_b128(&As[cur ^ 1][lrow * 32 + lcol], asrc + kb + 32);
            async_copy_b128(&Ws[cur ^ 1][lrow * 32 + lcol], wsrc + kb + 32);
            wait_async_le2();
        } else {
            wait_async_0();
        }
        __syncthreads();
        bf16x16 b  = load_fragB(Ws[cur], ni * 16, 0, 32);
        bf16x16 a0 = load_fragA(As[cur], mi0 * 16, 0, 32);
        acc[0] = wmma_bf16(a0, b, acc[0]);
        bf16x16 a1 = load_fragA(As[cur], (mi0 + 2) * 16, 0, 32);
        acc[1] = wmma_bf16(a1, b, acc[1]);
        __syncthreads();
    }
    for (int ts = 0; ts < 2; ++ts) {
        int mi = mi0 + ts * 2;
        for (int r = 0; r < 8; ++r) {
            int token = mb * 64 + mi * 16 + r + ((lane & 16) ? 8 : 0);
            int col   = nb * 64 + ni * 16 + (lane & 15);
            Out[(size_t)token * 1024 + col] = acc[ts][r] + bo[col];
        }
    }
}

// ---------------------------------------------------------------------------
// Host launcher
// ---------------------------------------------------------------------------
extern "C" void kernel_launch(void* const* d_in, const int* in_sizes, int n_in,
                              void* d_out, int out_size, void* d_ws, size_t ws_size,
                              hipStream_t stream) {
    (void)in_sizes; (void)n_in; (void)out_size; (void)ws_size;
    const float* X  = (const float*)d_in[0];   // [2048,1024]
    const float* Wq = (const float*)d_in[1];   // [1024,1024]
    const float* Wk = (const float*)d_in[2];
    const float* Wv = (const float*)d_in[3];
    const float* Wo = (const float*)d_in[4];
    const float* bo = (const float*)d_in[5];   // [1024]
    float* Out = (float*)d_out;

    // workspace carve-up (~50 MB)
    unsigned short* Xb  = (unsigned short*)d_ws;          // 2048*1024
    unsigned short* Wqb = Xb  + 2048 * 1024;              // 1024*1024 each
    unsigned short* Wkb = Wqb + 1024 * 1024;
    unsigned short* Wvb = Wkb + 1024 * 1024;
    unsigned short* Wob = Wvb + 1024 * 1024;
    float* Qf     = (float*)(Wob + 1024 * 1024);          // [16][2048][64]
    float* Kf     = Qf + (size_t)16 * 2048 * 64;
    float* Vf     = Kf + (size_t)16 * 2048 * 64;
    float* scales = Vf + (size_t)16 * 2048 * 64;          // 32 (pad 64)
    float* Ssum   = scales + 64;                          // [16][32][64*64]
    float* ksum   = Ssum + (size_t)16 * 32 * 4096;        // [16][32][64]
    float* vsum   = ksum + (size_t)16 * 32 * 64;
    unsigned short* attnb = (unsigned short*)(vsum + (size_t)16 * 32 * 64); // [2048][1024]

    const int total_cvt = (2048 + 4 * 1024) * 1024;       // 6M elements
    cvt_all<<<total_cvt / 256, 256, 0, stream>>>(X, Wq, Wk, Wv, Wo,
                                                 Xb, Wqb, Wkb, Wvb, Wob);

    gemm_qkv<<<dim3(32, 16, 3), 256, 0, stream>>>(Xb, Wqb, Wkb, Wvb, Qf, Kf, Vf);
    norm_scales<<<16, 256, 0, stream>>>(Qf, Kf, scales);
    chunk_sums<<<dim3(32, 16), 256, 0, stream>>>(Kf, Vf, scales, Ssum, ksum, vsum);
    chunk_prefix<<<16, 256, 0, stream>>>(Ssum, ksum, vsum);
    attn_chunk<<<dim3(32, 16), 256, 0, stream>>>(Qf, Kf, Vf, Ssum, ksum, vsum, scales, attnb);
    gemm_out<<<dim3(32, 16), 256, 0, stream>>>(attnb, Wob, bo, Out);
}
